// WordLevelAttention_14121852469424
// MI455X (gfx1250) — matrix operand order repaired and checked by reading, not compiled, not run on hardware
//
#include <hip/hip_runtime.h>

#define S_LEN 128
#define BATCH 1024
#define EMB   256
#define HID   256

typedef __attribute__((ext_vector_type(16))) __bf16 v16bf;
typedef __attribute__((ext_vector_type(8)))  float  v8f;

union FragU { uint4 u[2]; v16bf v; };

// 16-bit A-matrix 16x32 fragment (ISA 7.12.2): lane m = lane%16, half = lane/16.
// VGPR v (pairs): k = (v>=4?16:0) + half*8 + 2*(v%4)  -> two contiguous b128 loads.
__device__ __forceinline__ v16bf frag_a(const unsigned short* rowk, int half) {
  FragU f;
  f.u[0] = *reinterpret_cast<const uint4*>(rowk + half * 8);
  f.u[1] = *reinterpret_cast<const uint4*>(rowk + 16 + half * 8);
  return f.v;
}
// 16-bit B-matrix 32x16 fragment: lane n = lane%16; VGPR v: k = (v>=4?16:0) + half*8 + 2*(v%4).
// rowk points at weight row n (i.e. B[k,n] = W[n,k]) -> two contiguous b128 loads.
__device__ __forceinline__ v16bf frag_b(const unsigned short* rowk, int half) {
  FragU f;
  f.u[0] = *reinterpret_cast<const uint4*>(rowk + half * 8);
  f.u[1] = *reinterpret_cast<const uint4*>(rowk + 16 + half * 8);
  return f.v;
}

__device__ __forceinline__ unsigned short f2bf(float x) {
  unsigned u = __float_as_uint(x);
  u += 0x7FFFu + ((u >> 16) & 1u);          // round-to-nearest-even
  return (unsigned short)(u >> 16);
}

__device__ __forceinline__ float sigmoidf_(float x) {
  return 1.0f / (1.0f + __expf(-x));
}

// ---------------- prep kernels ----------------

__global__ void k_cvt_bf16(const float* __restrict__ src, unsigned short* __restrict__ dst, int n) {
  int i = blockIdx.x * blockDim.x + threadIdx.x;
  if (i < n) dst[i] = f2bf(src[i]);
}

__global__ void k_transpose_ww(const float* __restrict__ ww, unsigned short* __restrict__ wwT) {
  int i = blockIdx.x * blockDim.x + threadIdx.x;   // 512*512
  int r = i >> 9, c = i & 511;                     // wwT[r,c] = ww[c,r]
  wwT[i] = f2bf(ww[c * 512 + r]);
}

__global__ void k_gather(const int* __restrict__ tokens, const float* __restrict__ emb,
                         unsigned short* __restrict__ xbf) {
  int i  = blockIdx.x * blockDim.x + threadIdx.x;  // S*B*32 threads
  int t  = i >> 5;
  int e8 = (i & 31) * 8;
  int tok = tokens[t];
  const float* src = emb + (size_t)tok * EMB + e8;
  union { unsigned short h[8]; uint4 u; } pk;
#pragma unroll
  for (int j = 0; j < 8; ++j) pk.h[j] = f2bf(src[j]);
  *reinterpret_cast<uint4*>(xbf + (size_t)t * EMB + e8) = pk.u;
}

__global__ void k_init_h(const float* __restrict__ h0, float* __restrict__ hf,
                         unsigned short* __restrict__ hb) {
  int i = blockIdx.x * blockDim.x + threadIdx.x;   // 2*B*H
  int dir = i >> 18;                               // B*H = 262144
  int r   = i & 262143;
  float v = h0[i];
  size_t o = (size_t)(dir * 2) * 262144 + r;       // parity 0 slot
  hf[o] = v;
  hb[o] = f2bf(v);
}

// ---------------- GRU step: fused gate GEMM ----------------
// gates layout per (dir,b): [r(0:256)=x+h, z(256:512)=x+h, inn(512:768)=x only, hn(768:1024)=h only]
__global__ void k_gru_gemm(const unsigned short* __restrict__ xbf,
                           const unsigned short* __restrict__ wih,   // [2][768][256] bf16
                           const unsigned short* __restrict__ whh,   // [2][768][256] bf16
                           const unsigned short* __restrict__ hbf,   // [2][2][B][H] bf16
                           float* __restrict__ gates,                // [2][B][1024]
                           int step, int parity) {
  const int dir  = blockIdx.y;
  const int lane = threadIdx.x & 31;
  const int wave = threadIdx.x >> 5;
  const int tile = blockIdx.x * 8 + wave;          // 64x64 tiles
  const int mt = tile >> 6, nt = tile & 63;
  const int half = lane >> 4, mn = lane & 15;
  const int p = dir ? (S_LEN - 1 - step) : step;

  const unsigned short* xA = xbf + ((size_t)p * BATCH + mt * 16 + mn) * EMB;
  const unsigned short* hA = hbf + ((size_t)(dir * 2 + parity) * BATCH + mt * 16 + mn) * HID;

  v8f acc = {};
  if (nt < 32) {                                    // r,z: x*Wih + h*Whh fused (K=512)
    const unsigned short* wi = wih + ((size_t)dir * 768 + nt * 16 + mn) * EMB;
    const unsigned short* wh = whh + ((size_t)dir * 768 + nt * 16 + mn) * HID;
#pragma unroll
    for (int kt = 0; kt < 8; ++kt)
      acc = __builtin_amdgcn_wmma_f32_16x16x32_bf16(false, frag_a(xA + kt * 32, half),
                                                    false, frag_b(wi + kt * 32, half),
                                                    (short)0, acc, false, false);
#pragma unroll
    for (int kt = 0; kt < 8; ++kt)
      acc = __builtin_amdgcn_wmma_f32_16x16x32_bf16(false, frag_a(hA + kt * 32, half),
                                                    false, frag_b(wh + kt * 32, half),
                                                    (short)0, acc, false, false);
  } else if (nt < 48) {                             // inn: x*Wih rows 512..767
    const unsigned short* wi = wih + ((size_t)dir * 768 + 512 + (nt - 32) * 16 + mn) * EMB;
#pragma unroll
    for (int kt = 0; kt < 8; ++kt)
      acc = __builtin_amdgcn_wmma_f32_16x16x32_bf16(false, frag_a(xA + kt * 32, half),
                                                    false, frag_b(wi + kt * 32, half),
                                                    (short)0, acc, false, false);
  } else {                                          // hn: h*Whh rows 512..767
    const unsigned short* wh = whh + ((size_t)dir * 768 + 512 + (nt - 48) * 16 + mn) * HID;
#pragma unroll
    for (int kt = 0; kt < 8; ++kt)
      acc = __builtin_amdgcn_wmma_f32_16x16x32_bf16(false, frag_a(hA + kt * 32, half),
                                                    false, frag_b(wh + kt * 32, half),
                                                    (short)0, acc, false, false);
  }

  float* g = gates + ((size_t)dir * BATCH + mt * 16) * 1024 + nt * 16 + mn;
#pragma unroll
  for (int v = 0; v < 8; ++v) g[(size_t)(v + half * 8) * 1024] = acc[v];  // C/D: m = v + half*8, n = lane%16
}

// ---------------- GRU step: elementwise gate math ----------------
__global__ void k_gru_gate(const float* __restrict__ gates,
                           const float* __restrict__ bihf, const float* __restrict__ bhhf,
                           const float* __restrict__ bihb, const float* __restrict__ bhhb,
                           float* __restrict__ hf, unsigned short* __restrict__ hb,
                           float* __restrict__ fout, unsigned short* __restrict__ foutbf,
                           float* __restrict__ hout, int step, int parity) {
  const int dir = blockIdx.y;
  int i = blockIdx.x * blockDim.x + threadIdx.x;   // B*H
  int b = i >> 8, c = i & 255;
  const float* g  = gates + ((size_t)dir * BATCH + b) * 1024;
  const float* bi = dir ? bihb : bihf;
  const float* bh = dir ? bhhb : bhhf;

  float r  = sigmoidf_(g[c]        + bi[c]        + bh[c]);
  float z  = sigmoidf_(g[256 + c]  + bi[256 + c]  + bh[256 + c]);
  float nn = tanhf(g[512 + c] + bi[512 + c] + r * (g[768 + c] + bh[512 + c]));

  const size_t BH = (size_t)BATCH * HID;
  float hp = hf[(size_t)(dir * 2 + parity) * BH + (size_t)b * HID + c];
  float hnew = (1.0f - z) * nn + z * hp;

  size_t po = (size_t)(dir * 2 + (parity ^ 1)) * BH + (size_t)b * HID + c;
  hf[po] = hnew;
  hb[po] = f2bf(hnew);

  int p = dir ? (S_LEN - 1 - step) : step;         // backward output goes to original position
  size_t fo = ((size_t)p * BATCH + b) * 512 + dir * 256 + c;
  fout[fo]   = hnew;
  foutbf[fo] = f2bf(hnew);

  if (step == S_LEN - 1) hout[(size_t)dir * BH + (size_t)b * HID + c] = hnew;
}

// ---------------- attention: fused u = tanh(f@W+b), score = tanh(u@cw) ----------------
// One wave per 16-row tile. The full K=512 A-tile (16 fragments = 128 VGPRs) is
// preloaded into registers once; the n-tile loop then streams only the B operand
// (wwT, 512 KB, L2-resident) -> 32x less A traffic, no A-load waits in the WMMA chain.
__global__ void k_attn_score(const unsigned short* __restrict__ foutbf,
                             const unsigned short* __restrict__ wwT,   // [512][512] bf16, wwT[j,i]=ww[i,j]
                             const float* __restrict__ wbias, const float* __restrict__ cw,
                             float* __restrict__ scores) {
  int gid  = blockIdx.x * blockDim.x + threadIdx.x;
  int wid  = gid >> 5;                              // 8192 waves
  int lane = gid & 31;
  int half = lane >> 4, mn = lane & 15;

  const unsigned short* arow = foutbf + ((size_t)wid * 16 + mn) * 512;

  FragU af[16];                                     // entire 16x512 A tile in registers
#pragma unroll
  for (int kt = 0; kt < 16; ++kt) {
    af[kt].u[0] = *reinterpret_cast<const uint4*>(arow + kt * 32 + half * 8);
    af[kt].u[1] = *reinterpret_cast<const uint4*>(arow + kt * 32 + 16 + half * 8);
  }

  float partial[8];
#pragma unroll
  for (int v = 0; v < 8; ++v) partial[v] = 0.0f;

  for (int ntile = 0; ntile < 32; ++ntile) {
    const unsigned short* brow = wwT + ((size_t)ntile * 16 + mn) * 512;
    v8f acc = {};
#pragma unroll
    for (int kt = 0; kt < 16; ++kt)
      acc = __builtin_amdgcn_wmma_f32_16x16x32_bf16(false, af[kt].v,
                                                    false, frag_b(brow + kt * 32, half),
                                                    (short)0, acc, false, false);
    int n = ntile * 16 + mn;
    float cwn = cw[n], wbn = wbias[n];
#pragma unroll
    for (int v = 0; v < 8; ++v) partial[v] += tanhf(acc[v] + wbn) * cwn;
  }

  // reduce over the 16 N-lanes within each half (xor masks 1..8 keep halves separate)
#pragma unroll
  for (int v = 0; v < 8; ++v) {
    for (int m = 1; m < 16; m <<= 1) partial[v] += __shfl_xor(partial[v], m, 32);
  }
  if (mn == 0) {
#pragma unroll
    for (int v = 0; v < 8; ++v)
      scores[(size_t)wid * 16 + half * 8 + v] = tanhf(partial[v]);   // C/D row m = v + half*8
  }
}

__global__ void k_softmax(const float* __restrict__ scores, float* __restrict__ alpha) {
  int b = blockIdx.x * blockDim.x + threadIdx.x;
  if (b >= BATCH) return;
  float mx = -1e30f;
  for (int s = 0; s < S_LEN; ++s) mx = fmaxf(mx, scores[s * BATCH + b]);
  float sum = 0.0f;
  for (int s = 0; s < S_LEN; ++s) sum += __expf(scores[s * BATCH + b] - mx);
  float inv = 1.0f / sum;
  for (int s = 0; s < S_LEN; ++s) alpha[s * BATCH + b] = __expf(scores[s * BATCH + b] - mx) * inv;
}

__global__ void k_pool(const float* __restrict__ fout, const float* __restrict__ alpha,
                       float* __restrict__ out) {
  int i = blockIdx.x * blockDim.x + threadIdx.x;   // B*512
  int b = i >> 9, c = i & 511;
  float acc = 0.0f;
  for (int s = 0; s < S_LEN; ++s)
    acc += fout[((size_t)s * BATCH + b) * 512 + c] * alpha[s * BATCH + b];
  out[i] = acc;
}

// ---------------- launch ----------------

extern "C" void kernel_launch(void* const* d_in, const int* in_sizes, int n_in,
                              void* d_out, int out_size, void* d_ws, size_t ws_size,
                              hipStream_t stream) {
  (void)in_sizes; (void)n_in; (void)out_size; (void)ws_size;

  const int*   tokens = (const int*)d_in[0];
  const float* hidden = (const float*)d_in[1];
  const float* emb    = (const float*)d_in[2];
  const float* Wihf   = (const float*)d_in[3];
  const float* Whhf   = (const float*)d_in[4];
  const float* bihf   = (const float*)d_in[5];
  const float* bhhf   = (const float*)d_in[6];
  const float* Wihb   = (const float*)d_in[7];
  const float* Whhb   = (const float*)d_in[8];
  const float* bihb   = (const float*)d_in[9];
  const float* bhhb   = (const float*)d_in[10];
  const float* ww     = (const float*)d_in[11];
  const float* wbias  = (const float*)d_in[12];
  const float* cw     = (const float*)d_in[13];
  float* out = (float*)d_out;                       // [pooled: B*512][h_output: 2*B*H]

  char* ws = (char*)d_ws;
  size_t off = 0;
  auto alloc = [&](size_t bytes) -> void* {
    void* p = ws + off;
    off = (off + bytes + 255) & ~(size_t)255;
    return p;
  };
  unsigned short* xbf    = (unsigned short*)alloc((size_t)S_LEN * BATCH * EMB * 2);   // 64 MB
  unsigned short* wih    = (unsigned short*)alloc((size_t)2 * 768 * 256 * 2);
  unsigned short* whh    = (unsigned short*)alloc((size_t)2 * 768 * 256 * 2);
  unsigned short* wwT    = (unsigned short*)alloc((size_t)512 * 512 * 2);
  float*          gates  = (float*)alloc((size_t)2 * BATCH * 1024 * 4);
  float*          hf     = (float*)alloc((size_t)4 * BATCH * HID * 4);
  unsigned short* hb     = (unsigned short*)alloc((size_t)4 * BATCH * HID * 2);
  float*          fout   = (float*)alloc((size_t)S_LEN * BATCH * 512 * 4);            // 256 MB
  unsigned short* foutbf = (unsigned short*)alloc((size_t)S_LEN * BATCH * 512 * 2);   // 128 MB
  float*          scores = (float*)alloc((size_t)S_LEN * BATCH * 4);
  float*          alpha  = (float*)alloc((size_t)S_LEN * BATCH * 4);

  const int NW = 768 * 256;
  k_cvt_bf16<<<(NW + 255) / 256, 256, 0, stream>>>(Wihf, wih,      NW);
  k_cvt_bf16<<<(NW + 255) / 256, 256, 0, stream>>>(Wihb, wih + NW, NW);
  k_cvt_bf16<<<(NW + 255) / 256, 256, 0, stream>>>(Whhf, whh,      NW);
  k_cvt_bf16<<<(NW + 255) / 256, 256, 0, stream>>>(Whhb, whh + NW, NW);
  k_transpose_ww<<<(512 * 512) / 256, 256, 0, stream>>>(ww, wwT);
  k_gather<<<(S_LEN * BATCH * 32) / 256, 256, 0, stream>>>(tokens, emb, xbf);
  k_init_h<<<(2 * BATCH * HID) / 256, 256, 0, stream>>>(hidden, hf, hb);

  for (int s = 0; s < S_LEN; ++s) {
    k_gru_gemm<<<dim3(512, 2), 256, 0, stream>>>(xbf, wih, whh, hb, gates, s, s & 1);
    k_gru_gate<<<dim3((BATCH * HID) / 256, 2), 256, 0, stream>>>(
        gates, bihf, bhhf, bihb, bhhb, hf, hb, fout, foutbf,
        out + (size_t)BATCH * 512, s, s & 1);
  }

  k_attn_score<<<(8192 * 32) / 256, 256, 0, stream>>>(foutbf, wwT, wbias, cw, scores);
  k_softmax<<<4, 256, 0, stream>>>(scores, alpha);
  k_pool<<<(BATCH * 512) / 256, 256, 0, stream>>>(fout, alpha, out);
}
